// CausalSelfAttention_2516850835841
// MI455X (gfx1250) — matrix operand order repaired and checked by reading, not compiled, not run on hardware
//
#include <hip/hip_runtime.h>
#include <hip/hip_bf16.h>

typedef __bf16 bf16;
typedef __attribute__((ext_vector_type(8)))  __bf16 v8bf;
typedef __attribute__((ext_vector_type(16))) __bf16 v16bf;
typedef __attribute__((ext_vector_type(8)))  float  v8f;

#define C_DIM 1024
#define T_DIM 2048
#define B_DIM 4
#define H_DIM 16
#define HS    64
#define C3    3072
#define M_DIM 8192   // B*T

// GEMM block tiling
#define BM 128
#define BN 64
#define BK 64
#define LDP 72       // padded LDS row stride (elements) to de-stagger banks

// ---------------- async copy-to-LDS (guarded, with sync fallback) ----------------

#if defined(__has_builtin)
#if __has_builtin(__builtin_amdgcn_global_load_async_to_lds_b128)
#define HAS_ASYNC_LDS 1
#endif
#endif
#ifndef HAS_ASYNC_LDS
#define HAS_ASYNC_LDS 0
#endif

// pointee type revealed by the round-2 diagnostic: int __attribute__((vector_size(16)))
typedef int i32x4 __attribute__((vector_size(16)));
typedef __attribute__((address_space(1))) i32x4* as1_i32x4p;
typedef __attribute__((address_space(3))) i32x4* as3_i32x4p;

// copy 16 bytes (8 bf16) global -> LDS, per lane
__device__ __forceinline__ void copy16(const bf16* g, bf16* l) {
#if HAS_ASYNC_LDS
  __builtin_amdgcn_global_load_async_to_lds_b128(
      (as1_i32x4p)(uintptr_t)g,
      (as3_i32x4p)(unsigned int)(uintptr_t)l,
      /*offset=*/0, /*cpol=*/0);
#else
  *(v8bf*)l = *(const v8bf*)g;
#endif
}

__device__ __forceinline__ void async_wait0() {
#if HAS_ASYNC_LDS
#if __has_builtin(__builtin_amdgcn_s_wait_asynccnt)
  __builtin_amdgcn_s_wait_asynccnt(0);
#else
  asm volatile("s_wait_asynccnt 0" ::: "memory");
#endif
#endif
}

// ---------------- fragment helpers (CDNA5 WMMA 16x16x32 bf16) ----------------

__device__ __forceinline__ v16bf pack16(v8bf lo, v8bf hi) {
  union { v16bf v; v8bf h[2]; } u;
  u.h[0] = lo; u.h[1] = hi;
  return u.v;
}

// A-matrix 16x32 (MxK): lane holds row = lane%16; half = lane/16 selects
// K runs {k0 + h*8 .. +7} and {k0 + 16 + h*8 .. +7}.
__device__ __forceinline__ v16bf load_frag_a(const bf16* rowptr, int k0, int half) {
  v8bf lo = *(const v8bf*)(rowptr + k0 + half * 8);
  v8bf hi = *(const v8bf*)(rowptr + k0 + 16 + half * 8);
  return pack16(lo, hi);
}

// B-matrix 32x16 (KxN), stored N-major (rowptr = column N of B, contiguous in K):
// lane holds col = lane%16; K runs {k0 + h*8 .. +7} and {k0 + 16 + h*8 .. +7}.
__device__ __forceinline__ v16bf load_frag_b(const bf16* rowptr, int k0, int half) {
  const bf16* p = rowptr + k0 + half * 8;
  return pack16(*(const v8bf*)p, *(const v8bf*)(p + 16));
}

__device__ __forceinline__ v8f wmma_bf16(v16bf a, v16bf b, v8f c) {
  return __builtin_amdgcn_wmma_f32_16x16x32_bf16(
      /*neg_a=*/false, a, /*neg_b=*/false, b,
      /*c_mod=*/(short)0, c, /*reuse_a=*/false, /*reuse_b=*/false);
}

// ---------------- staging kernels (fp32 -> bf16, optional transpose) ----------------

__global__ void convert_bf16_kernel(const float* __restrict__ in, bf16* __restrict__ out, int n) {
  int i = blockIdx.x * blockDim.x + threadIdx.x;
  if (i < n) out[i] = (bf16)in[i];
}

// in: [rows, cols] fp32 -> out: [cols, rows] bf16
__global__ void transpose_bf16_kernel(const float* __restrict__ in, bf16* __restrict__ out,
                                      int rows, int cols) {
  int i = blockIdx.x * blockDim.x + threadIdx.x;
  if (i < rows * cols) {
    int r = i / cols, c = i - r * cols;
    out[c * rows + r] = (bf16)in[i];
  }
}

// ---------------- QKV GEMM: [8192,1024] @ [1024,3072] + bias ----------------
// 256 threads = 8 waves; block tile 128x64 (waves 4x2, each 32x32), K chunks of 64
// double-buffered through LDS via async copy. Epilogue scatters to q/k (head
// layout) and transposed v; q gets *0.125 (hs^-0.5) folded in.

__global__ void __launch_bounds__(256)
gemm_qkv_kernel(const bf16* __restrict__ x, const bf16* __restrict__ wT,
                const float* __restrict__ bias,
                bf16* __restrict__ qb, bf16* __restrict__ kb,
                bf16* __restrict__ vT) {
  __shared__ bf16 lA[2][BM * LDP];
  __shared__ bf16 lB[2][BN * LDP];

  const int NBLK = C3 / BN;  // 48
  int bm = blockIdx.x / NBLK, bn = blockIdx.x - bm * NBLK;
  int m0 = bm * BM, n0 = bn * BN;
  int tid = threadIdx.x;
  int lane = tid & 31, wave = tid >> 5;
  int half = lane >> 4, col = lane & 15;
  int wm = wave >> 1, wn = wave & 1;

  const bf16* Abase = x + (size_t)m0 * C_DIM;
  const bf16* Bbase = wT + (size_t)n0 * C_DIM;

  auto copy_chunk = [&](int buf, int k0) {
    #pragma unroll
    for (int u = 0; u < 4; u++) {           // A: 128x64 = 1024 16B units
      int unit = tid + u * 256;
      int row = unit >> 3, kc = (unit & 7) * 8;
      copy16(Abase + (size_t)row * C_DIM + k0 + kc, &lA[buf][row * LDP + kc]);
    }
    #pragma unroll
    for (int u = 0; u < 2; u++) {           // B: 64x64 = 512 16B units
      int unit = tid + u * 256;
      int row = unit >> 3, kc = (unit & 7) * 8;
      copy16(Bbase + (size_t)row * C_DIM + k0 + kc, &lB[buf][row * LDP + kc]);
    }
  };

  v8f c00 = {}, c01 = {}, c10 = {}, c11 = {};

  copy_chunk(0, 0);
  async_wait0();
  __syncthreads();

  const int KT = C_DIM / BK;  // 16
  for (int kc = 0; kc < KT; kc++) {
    int cur = kc & 1;
    if (kc + 1 < KT) copy_chunk(1 - cur, (kc + 1) * BK);

    const bf16* a0p = &lA[cur][(wm * 32 + col) * LDP];
    const bf16* a1p = &lA[cur][(wm * 32 + 16 + col) * LDP];
    const bf16* b0p = &lB[cur][(wn * 32 + col) * LDP];
    const bf16* b1p = &lB[cur][(wn * 32 + 16 + col) * LDP];
    #pragma unroll
    for (int ks = 0; ks < BK; ks += 32) {
      v16bf a0 = load_frag_a(a0p, ks, half);
      v16bf a1 = load_frag_a(a1p, ks, half);
      v16bf b0 = load_frag_b(b0p, ks, half);
      v16bf b1 = load_frag_b(b1p, ks, half);
      c00 = wmma_bf16(a0, b0, c00);
      c01 = wmma_bf16(a0, b1, c01);
      c10 = wmma_bf16(a1, b0, c10);
      c11 = wmma_bf16(a1, b1, c11);
    }
    async_wait0();
    __syncthreads();
  }

  // epilogue: wave tile at (m0 + wm*32, n0 + wn*32); C-layout row = half*8+r
  int m0w = m0 + wm * 32, n0w = n0 + wn * 32;
  v8f cc[4] = {c00, c01, c10, c11};
  #pragma unroll
  for (int s = 0; s < 4; s++) {
    int rowbase = m0w + (s >> 1) * 16;
    int n = n0w + ((s & 1) * 16) + col;
    float bv = bias[n];
    int which = n >> 10;          // 0=q 1=k 2=v
    int hh = (n & 1023) >> 6;     // head
    int d  = n & 63;
    #pragma unroll
    for (int r = 0; r < 8; r++) {
      int row = rowbase + half * 8 + r;
      int b = row >> 11, t = row & (T_DIM - 1);
      float v = cc[s][r] + bv;
      size_t bh = (size_t)(b * H_DIM + hh);
      if (which == 0) {
        qb[(bh * T_DIM + t) * HS + d] = (bf16)(v * 0.125f);
      } else if (which == 1) {
        kb[(bh * T_DIM + t) * HS + d] = (bf16)v;
      } else {
        vT[(bh * HS + d) * T_DIM + t] = (bf16)v;
      }
    }
  }
}

// ---------------- flash attention: one wave = 16 queries of one (b,h) ----------------

__global__ void __launch_bounds__(128)
attn_kernel(const bf16* __restrict__ q, const bf16* __restrict__ k,
            const bf16* __restrict__ vT, bf16* __restrict__ y) {
  __shared__ bf16 plds[4][16 * 32];  // per-wave P staging (C-layout -> A-layout)
  int wib  = threadIdx.x >> 5;
  int lane = threadIdx.x & 31;
  int wave = blockIdx.x * 4 + wib;
  int qt = wave & 127;       // T/16 = 128 query tiles
  int bh = wave >> 7;        // b*H + h
  int q0 = qt * 16;
  int half = lane >> 4, col = lane & 15;

  const bf16* qp = q  + (size_t)bh * T_DIM * HS;
  const bf16* kp = k  + (size_t)bh * T_DIM * HS;
  const bf16* vp = vT + (size_t)bh * HS * T_DIM;

  const bf16* qrow = qp + (size_t)(q0 + col) * HS;
  v16bf qa0 = load_frag_a(qrow, 0, half);
  v16bf qa1 = load_frag_a(qrow, 32, half);

  float mrow[8], lrow[8];
  #pragma unroll
  for (int r = 0; r < 8; r++) { mrow[r] = -__builtin_inff(); lrow[r] = 0.f; }
  v8f o0 = {}, o1 = {}, o2 = {}, o3 = {};

  bf16* pl = plds[wib];

  for (int key0 = 0; key0 <= q0 + 15; key0 += 32) {
    // ---- S = q @ k^T over this 32-key tile (two 16-col halves) ----
    const bf16* kr0 = kp + (size_t)(key0 + col) * HS;
    const bf16* kr1 = kp + (size_t)(key0 + 16 + col) * HS;
    v8f s0 = {}, s1 = {};
    s0 = wmma_bf16(qa0, load_frag_b(kr0, 0, half), s0);
    s0 = wmma_bf16(qa1, load_frag_b(kr0, 32, half), s0);
    s1 = wmma_bf16(qa0, load_frag_b(kr1, 0, half), s1);
    s1 = wmma_bf16(qa1, load_frag_b(kr1, 32, half), s1);

    // ---- causal mask + online softmax (rows live per 16-lane half) ----
    float p0[8], p1[8], scl[8];
    #pragma unroll
    for (int r = 0; r < 8; r++) {
      int qrowi = q0 + half * 8 + r;
      bool v0ok = (key0 + col) <= qrowi;
      bool v1ok = (key0 + 16 + col) <= qrowi;
      float a = v0ok ? s0[r] : -__builtin_inff();
      float b = v1ok ? s1[r] : -__builtin_inff();
      float t = fmaxf(a, b);
      #pragma unroll
      for (int off = 1; off < 16; off <<= 1) t = fmaxf(t, __shfl_xor(t, off, 32));
      float mn = fmaxf(mrow[r], t);
      scl[r] = __expf(mrow[r] - mn);  // exp(-inf)=0 on first pass
      mrow[r] = mn;
      p0[r] = v0ok ? __expf(s0[r] - mn) : 0.f;
      p1[r] = v1ok ? __expf(s1[r] - mn) : 0.f;
      float sm = p0[r] + p1[r];
      #pragma unroll
      for (int off = 1; off < 16; off <<= 1) sm += __shfl_xor(sm, off, 32);
      lrow[r] = lrow[r] * scl[r] + sm;
    }
    #pragma unroll
    for (int r = 0; r < 8; r++) {
      o0[r] *= scl[r]; o1[r] *= scl[r]; o2[r] *= scl[r]; o3[r] *= scl[r];
    }

    // ---- stage P (16x32 bf16) through LDS: C-layout -> A-layout ----
    #pragma unroll
    for (int r = 0; r < 8; r++) {
      int m = half * 8 + r;
      pl[m * 32 + col]      = (bf16)p0[r];
      pl[m * 32 + 16 + col] = (bf16)p1[r];
    }
    v16bf pa;
    {
      int m = col;  // A row = lane%16
      v8bf x0 = *(const v8bf*)(pl + m * 32 + half * 8);
      v8bf x1 = *(const v8bf*)(pl + m * 32 + 16 + half * 8);
      pa = pack16(x0, x1);
    }

    // ---- O += P @ V (vT rows are contiguous in t = K dim) ----
    o0 = wmma_bf16(pa, load_frag_b(vp + (size_t)(0 * 16 + col) * T_DIM, key0, half), o0);
    o1 = wmma_bf16(pa, load_frag_b(vp + (size_t)(1 * 16 + col) * T_DIM, key0, half), o1);
    o2 = wmma_bf16(pa, load_frag_b(vp + (size_t)(2 * 16 + col) * T_DIM, key0, half), o2);
    o3 = wmma_bf16(pa, load_frag_b(vp + (size_t)(3 * 16 + col) * T_DIM, key0, half), o3);
  }

  // ---- finalize: O /= l, write y[b][t][h*64+d] as bf16 ----
  int b = bh >> 4, h = bh & 15;
  #pragma unroll
  for (int r = 0; r < 8; r++) {
    float inv = 1.f / lrow[r];
    int t = q0 + half * 8 + r;
    size_t base = ((size_t)(b * T_DIM + t)) * C_DIM + h * HS;
    y[base + 0 * 16 + col] = (bf16)(o0[r] * inv);
    y[base + 1 * 16 + col] = (bf16)(o1[r] * inv);
    y[base + 2 * 16 + col] = (bf16)(o2[r] * inv);
    y[base + 3 * 16 + col] = (bf16)(o3[r] * inv);
  }
}

// ---------------- projection GEMM: [8192,1024] @ [1024,1024] + bias -> fp32 ----------------

__global__ void __launch_bounds__(256)
gemm_proj_kernel(const bf16* __restrict__ y, const bf16* __restrict__ wT,
                 const float* __restrict__ bias, float* __restrict__ out) {
  __shared__ bf16 lA[2][BM * LDP];
  __shared__ bf16 lB[2][BN * LDP];

  const int NBLK = C_DIM / BN;  // 16
  int bm = blockIdx.x / NBLK, bn = blockIdx.x - bm * NBLK;
  int m0 = bm * BM, n0 = bn * BN;
  int tid = threadIdx.x;
  int lane = tid & 31, wave = tid >> 5;
  int half = lane >> 4, col = lane & 15;
  int wm = wave >> 1, wn = wave & 1;

  const bf16* Abase = y + (size_t)m0 * C_DIM;
  const bf16* Bbase = wT + (size_t)n0 * C_DIM;

  auto copy_chunk = [&](int buf, int k0) {
    #pragma unroll
    for (int u = 0; u < 4; u++) {
      int unit = tid + u * 256;
      int row = unit >> 3, kc = (unit & 7) * 8;
      copy16(Abase + (size_t)row * C_DIM + k0 + kc, &lA[buf][row * LDP + kc]);
    }
    #pragma unroll
    for (int u = 0; u < 2; u++) {
      int unit = tid + u * 256;
      int row = unit >> 3, kc = (unit & 7) * 8;
      copy16(Bbase + (size_t)row * C_DIM + k0 + kc, &lB[buf][row * LDP + kc]);
    }
  };

  v8f c00 = {}, c01 = {}, c10 = {}, c11 = {};

  copy_chunk(0, 0);
  async_wait0();
  __syncthreads();

  const int KT = C_DIM / BK;  // 16
  for (int kc = 0; kc < KT; kc++) {
    int cur = kc & 1;
    if (kc + 1 < KT) copy_chunk(1 - cur, (kc + 1) * BK);

    const bf16* a0p = &lA[cur][(wm * 32 + col) * LDP];
    const bf16* a1p = &lA[cur][(wm * 32 + 16 + col) * LDP];
    const bf16* b0p = &lB[cur][(wn * 32 + col) * LDP];
    const bf16* b1p = &lB[cur][(wn * 32 + 16 + col) * LDP];
    #pragma unroll
    for (int ks = 0; ks < BK; ks += 32) {
      v16bf a0 = load_frag_a(a0p, ks, half);
      v16bf a1 = load_frag_a(a1p, ks, half);
      v16bf b0 = load_frag_b(b0p, ks, half);
      v16bf b1 = load_frag_b(b1p, ks, half);
      c00 = wmma_bf16(a0, b0, c00);
      c01 = wmma_bf16(a0, b1, c01);
      c10 = wmma_bf16(a1, b0, c10);
      c11 = wmma_bf16(a1, b1, c11);
    }
    async_wait0();
    __syncthreads();
  }

  int m0w = m0 + wm * 32, n0w = n0 + wn * 32;
  v8f cc[4] = {c00, c01, c10, c11};
  #pragma unroll
  for (int s = 0; s < 4; s++) {
    int rowbase = m0w + (s >> 1) * 16;
    int n = n0w + ((s & 1) * 16) + col;
    float bv = bias[n];
    #pragma unroll
    for (int r = 0; r < 8; r++) {
      int row = rowbase + half * 8 + r;
      out[(size_t)row * C_DIM + n] = cc[s][r] + bv;
    }
  }
}

// ---------------- launch ----------------

extern "C" void kernel_launch(void* const* d_in, const int* in_sizes, int n_in,
                              void* d_out, int out_size, void* d_ws, size_t ws_size,
                              hipStream_t stream) {
  const float* x      = (const float*)d_in[0];  // [4,2048,1024]
  const float* w_attn = (const float*)d_in[1];  // [1024,3072]
  const float* b_attn = (const float*)d_in[2];  // [3072]
  const float* w_proj = (const float*)d_in[3];  // [1024,1024]
  const float* b_proj = (const float*)d_in[4];  // [1024]
  float* out = (float*)d_out;

  char* ws = (char*)d_ws;
  size_t off = 0;
  auto alloc = [&](size_t bytes) -> void* {
    void* p = ws + off;
    off += (bytes + 255) & ~(size_t)255;
    return p;
  };
  bf16* xb  = (bf16*)alloc((size_t)M_DIM * C_DIM * 2);   // 16 MB
  bf16* wTa = (bf16*)alloc((size_t)C3 * C_DIM * 2);      //  6 MB  [3072,1024]
  bf16* wTp = (bf16*)alloc((size_t)C_DIM * C_DIM * 2);   //  2 MB  [1024,1024]
  bf16* qb  = (bf16*)alloc((size_t)B_DIM * H_DIM * T_DIM * HS * 2);  // 16 MB
  bf16* kb  = (bf16*)alloc((size_t)B_DIM * H_DIM * T_DIM * HS * 2);  // 16 MB
  bf16* vT  = (bf16*)alloc((size_t)B_DIM * H_DIM * HS * T_DIM * 2);  // 16 MB
  bf16* yb  = (bf16*)alloc((size_t)M_DIM * C_DIM * 2);   // 16 MB

  int nx = M_DIM * C_DIM;
  convert_bf16_kernel<<<(nx + 255) / 256, 256, 0, stream>>>(x, xb, nx);
  transpose_bf16_kernel<<<(C_DIM * C3 + 255) / 256, 256, 0, stream>>>(w_attn, wTa, C_DIM, C3);
  transpose_bf16_kernel<<<(C_DIM * C_DIM + 255) / 256, 256, 0, stream>>>(w_proj, wTp, C_DIM, C_DIM);

  // QKV GEMM: (8192/128) x (3072/64) = 64 x 48 = 3072 blocks
  gemm_qkv_kernel<<<3072, 256, 0, stream>>>(xb, wTa, b_attn, qb, kb, vT);

  // attention: B*H*(T/16) = 8192 waves, 4 waves/block
  attn_kernel<<<8192 / 4, 128, 0, stream>>>(qb, kb, vT, yb);

  // proj GEMM: (8192/128) x (1024/64) = 64 x 16 = 1024 blocks
  gemm_proj_kernel<<<1024, 256, 0, stream>>>(yb, wTp, b_proj, out);
}
